// CPGNN_16939351015663
// MI455X (gfx1250) — matrix operand order repaired and checked by reading, not compiled, not run on hardware
//
#include <hip/hip_runtime.h>
#include <hip/hip_bf16.h>

// Problem constants (match reference)
constexpr int kN = 10000;   // nodes
constexpr int kF = 512;     // F_IN
constexpr int kH = 256;     // HID
constexpr int kC = 16;      // classes
constexpr int kIters = 4;   // N_POST_ITER
constexpr int kSplit = 8;   // K-splits for the Wd=16 adjacency GEMMs

typedef __attribute__((ext_vector_type(16))) __bf16 v16bf;
typedef __attribute__((ext_vector_type(8)))  __bf16 v8bf;
typedef __attribute__((ext_vector_type(8)))  float  v8f;
typedef __attribute__((ext_vector_type(4)))  float  v4f;

// ---------------------------------------------------------------------------
// WMMA GEMM:  out = epi( A[M,K] (fp32 row-major) @ Bt^T )
//   Bt: B stored TRANSPOSED bf16 [n][k] (row stride K) -> 32B fragment loads.
//   A streams fp32 (4x float4/lane), cvt->bf16 in regs, fp32 accumulation.
//   Register double-buffering: k-tile i+1 loads issue while tile i computes.
//
//   NT=4 : 4 waves/block cover a 256-wide row panel (one m-tile/block).
//   NT=1 : 4 waves/block cover 4 m-tiles of a 16-wide output.
//   SPLIT: grid.y k-splits; each writes a raw fp32 partial [M,Wd] tile to
//          Out + blockIdx.y*M*Wd; epilogue deferred to reduce_split().
// ---------------------------------------------------------------------------
template<int NT, bool RELU, bool SPLIT>
__global__ __launch_bounds__(128) void wmma_mm(
    const float*  __restrict__ A, const __bf16* __restrict__ Bt,
    float* __restrict__ Out, __bf16* __restrict__ OutT,
    int M, int K, int Wd, int ktPer,
    const float* __restrict__ bias,   // [Wd] or nullptr   (unsplit only)
    const float* __restrict__ addM,   // [M,Wd] or nullptr (unsplit only)
    float scalarAdd)
{
  const int lane = threadIdx.x & 31;
  const int wave = threadIdx.x >> 5;
  const int r16  = lane & 15;
  const int half = lane >> 4;

  int mtile, n0base;
  if (NT == 4) { mtile = blockIdx.x;            n0base = wave * (NT * 16); }
  else         { mtile = blockIdx.x * 4 + wave; n0base = 0;                }
  const int m0 = mtile * 16;
  if (m0 >= M) return;                       // wave-uniform: EXEC stays all-1s

  const int ktTotal = (K + 31) >> 5;         // #k-tiles incl. partial
  const int ktFull  = K >> 5;                // #full 32-wide k-tiles
  int ktStart, ktEnd;
  if (SPLIT) {
    ktStart = blockIdx.y * ktPer;
    ktEnd   = ktStart + ktPer; if (ktEnd > ktTotal) ktEnd = ktTotal;
    if (ktStart > ktTotal) ktStart = ktTotal;
  } else { ktStart = 0; ktEnd = ktTotal; }
  const int  fullEnd = (ktEnd < ktFull) ? ktEnd : ktFull;
  const int  nFull   = fullEnd - ktStart;
  const bool hasTail = ((K & 31) != 0) && (ktEnd == ktTotal) && (ktStart < ktTotal);

  v8f acc[NT];
#pragma unroll
  for (int t = 0; t < NT; ++t) acc[t] = v8f{0.f,0.f,0.f,0.f,0.f,0.f,0.f,0.f};

  const float* arow = A + (size_t)(m0 + r16) * (size_t)K;
  const __bf16* brow[NT];
#pragma unroll
  for (int t = 0; t < NT; ++t)
    brow[t] = Bt + (size_t)(n0base + t * 16 + r16) * (size_t)K + half * 16;

  // Fragment layouts used here (wave32; contiguous k, not the ISA table order):
  //  A 16x32: lane&15=M; elem e -> K = k0 + e + 16*half
  //  B 32x16: lane&15=N; elem e -> K = k0 + e + 16*half
  //  C/D    : lane&15=N; vgpr v -> M = v + 8*half
  auto loadFrag = [&](int k0, v4f& a0, v4f& a1, v4f& a2, v4f& a3,
                      v16bf (&bf)[NT]) {
    const v4f* p = reinterpret_cast<const v4f*>(arow + k0 + half * 16);
    a0 = p[0]; a1 = p[1];
    a2 = p[2]; a3 = p[3];
#pragma unroll
    for (int t = 0; t < NT; ++t)
      bf[t] = *reinterpret_cast<const v16bf*>(brow[t] + k0);
  };
  auto computeTile = [&](const v4f& a0, const v4f& a1, const v4f& a2,
                         const v4f& a3, const v16bf (&bf)[NT]) {
    v16bf afrag;
#pragma unroll
    for (int e = 0; e < 4; ++e) {
      afrag[e]      = (__bf16)a0[e];
      afrag[e + 4]  = (__bf16)a1[e];
      afrag[e + 8]  = (__bf16)a2[e];
      afrag[e + 12] = (__bf16)a3[e];
    }
#pragma unroll
    for (int t = 0; t < NT; ++t)
      acc[t] = __builtin_amdgcn_wmma_f32_16x16x32_bf16(
          false, afrag, false, bf[t], (short)0, acc[t], false, false);
  };

  if (nFull > 0) {                           // software-pipelined main loop
    const int k0base = ktStart * 32;
    v4f a0, a1, a2, a3; v16bf bf[NT];
    loadFrag(k0base, a0, a1, a2, a3, bf);
    for (int i = 1; i < nFull; ++i) {
      v4f c0 = a0, c1 = a1, c2 = a2, c3 = a3;
      v16bf cb[NT];
#pragma unroll
      for (int t = 0; t < NT; ++t) cb[t] = bf[t];
      loadFrag(k0base + i * 32, a0, a1, a2, a3, bf);   // prefetch next tile
      computeTile(c0, c1, c2, c3, cb);                 // compute current
    }
    computeTile(a0, a1, a2, a3, bf);                   // last tile
  }

  if (hasTail) {                             // guarded tail (10000 = 32*312+16)
    const int k0 = ktFull * 32;
    v16bf afrag;
#pragma unroll
    for (int e = 0; e < 8; ++e) {
      const int ka = k0 + half * 16 + e;
      const int kb = ka + 8;
      afrag[e]     = (ka < K) ? (__bf16)arow[ka] : (__bf16)0.0f;
      afrag[e + 8] = (kb < K) ? (__bf16)arow[kb] : (__bf16)0.0f;
    }
#pragma unroll
    for (int t = 0; t < NT; ++t) {
      v16bf bfrag;
#pragma unroll
      for (int e = 0; e < 16; ++e) {
        const int kk = k0 + half * 16 + e;
        bfrag[e] = (kk < K) ? brow[t][k0 + e] : (__bf16)0.0f;
      }
      acc[t] = __builtin_amdgcn_wmma_f32_16x16x32_bf16(
          false, afrag, false, bfrag, (short)0, acc[t], false, false);
    }
  }

  if (SPLIT) {                               // raw fp32 partial tile
    float* P = Out + (size_t)blockIdx.y * (size_t)M * Wd;
#pragma unroll
    for (int t = 0; t < NT; ++t) {
      const int n = n0base + t * 16 + r16;
#pragma unroll
      for (int v = 0; v < 8; ++v)
        P[(size_t)(m0 + v + half * 8) * Wd + n] = acc[t][v];
    }
  } else {                                   // fused epilogue
#pragma unroll
    for (int t = 0; t < NT; ++t) {
      const int n  = n0base + t * 16 + r16;
      const float bn = bias ? bias[n] : 0.0f;
      float val[8];
#pragma unroll
      for (int v = 0; v < 8; ++v) {
        const int m = m0 + v + half * 8;
        float x = acc[t][v] + bn + scalarAdd;
        if (addM) x += addM[(size_t)m * Wd + n];
        if (RELU) x = fmaxf(x, 0.0f);
        val[v] = x;
        if (Out) Out[(size_t)m * Wd + n] = x; // coalesced b32 stores
      }
      if (OutT) {                             // one aligned 16B bf16 store
        v8bf ot;
#pragma unroll
        for (int v = 0; v < 8; ++v) ot[v] = (__bf16)val[v];
        *reinterpret_cast<v8bf*>(OutT + (size_t)n * M + m0 + half * 8) = ot;
      }
    }
  }
}

// ---------------------------------------------------------------------------
// Out[i,c] = sum_s P[s][i][c] + bias[c] + addM[i,c] + scalarAdd
// Deterministic K-split reduction (no float atomics).
// ---------------------------------------------------------------------------
__global__ __launch_bounds__(256) void reduce_split(
    const float* __restrict__ P, float* __restrict__ Out,
    const float* __restrict__ bias, const float* __restrict__ addM,
    float scalarAdd, int M)
{
  const int idx = blockIdx.x * 256 + threadIdx.x;        // over M*kC
  if (idx >= M * kC) return;
  float s = scalarAdd;
#pragma unroll
  for (int p = 0; p < kSplit; ++p) s += P[(size_t)p * M * kC + idx];
  if (bias) s += bias[idx & (kC - 1)];
  if (addM) s += addM[idx];
  Out[idx] = s;
}

// ---------------------------------------------------------------------------
// out[c][r] = (bf16) in[r][c]   (tiny weight transposes; coalesced stores)
// ---------------------------------------------------------------------------
__global__ __launch_bounds__(256) void transpose_cvt(
    const float* __restrict__ in, __bf16* __restrict__ out, int R, int Cw)
{
  const int idx = blockIdx.x * 256 + threadIdx.x;
  if (idx >= R * Cw) return;
  const int c = idx / R, r = idx - c * R;     // consecutive idx -> contig out
  out[(size_t)c * R + r] = (__bf16)in[(size_t)r * Cw + c];
}

// ---------------------------------------------------------------------------
// Tt[c][i] = (bf16) ( S[i,:] @ Hm )[c]    (S:[M,16] f32, Hm:[16,16] f32)
// ---------------------------------------------------------------------------
__global__ __launch_bounds__(256) void small_mm_t(
    const float* __restrict__ S, const float* __restrict__ Hm,
    __bf16* __restrict__ Tt, int M)
{
  __shared__ float Hs[kC * kC];
  Hs[threadIdx.x] = Hm[threadIdx.x & (kC * kC - 1)];
  __syncthreads();
  const int i = blockIdx.x * 256 + threadIdx.x;
  if (i >= M) return;
  const v4f* sr = reinterpret_cast<const v4f*>(S + (size_t)i * kC);
  v4f s0 = sr[0], s1 = sr[1], s2 = sr[2], s3 = sr[3];
  float s[kC];
#pragma unroll
  for (int e = 0; e < 4; ++e) {
    s[e] = s0[e]; s[4 + e] = s1[e]; s[8 + e] = s2[e]; s[12 + e] = s3[e];
  }
#pragma unroll
  for (int c = 0; c < kC; ++c) {
    float acc = 0.0f;
#pragma unroll
    for (int k = 0; k < kC; ++k) acc = fmaf(s[k], Hs[k * kC + c], acc);
    Tt[(size_t)c * M + i] = (__bf16)acc;     // coalesced b16 stores
  }
}

// ---------------------------------------------------------------------------
// E[i,:] = softmax(L[i,:]) - 1/C
// ---------------------------------------------------------------------------
__global__ __launch_bounds__(256) void softmax_e(
    const float* __restrict__ L, float* __restrict__ E, int M, float invC)
{
  const int i = blockIdx.x * 256 + threadIdx.x;
  if (i >= M) return;
  const v4f* lr = reinterpret_cast<const v4f*>(L + (size_t)i * kC);
  v4f l0 = lr[0], l1 = lr[1], l2 = lr[2], l3 = lr[3];
  float v[kC];
#pragma unroll
  for (int e = 0; e < 4; ++e) {
    v[e] = l0[e]; v[4 + e] = l1[e]; v[8 + e] = l2[e]; v[12 + e] = l3[e];
  }
  float mx = v[0];
#pragma unroll
  for (int c = 1; c < kC; ++c) mx = fmaxf(mx, v[c]);
  float sum = 0.0f;
#pragma unroll
  for (int c = 0; c < kC; ++c) { v[c] = __expf(v[c] - mx); sum += v[c]; }
  const float r = 1.0f / sum;
#pragma unroll
  for (int c = 0; c < kC; ++c) E[(size_t)i * kC + c] = v[c] * r - invC;
}

// ---------------------------------------------------------------------------
extern "C" void kernel_launch(void* const* d_in, const int* in_sizes, int n_in,
                              void* d_out, int out_size, void* d_ws, size_t ws_size,
                              hipStream_t stream)
{
  (void)in_sizes; (void)n_in; (void)out_size; (void)ws_size;
  const float* raw_adj = (const float*)d_in[0];
  const float* normed  = (const float*)d_in[1];
  const float* feat    = (const float*)d_in[2];
  // d_in[3] = y_onehot, d_in[4] = train_mask: unused by the reference output
  const float* W1 = (const float*)d_in[5];
  const float* b1 = (const float*)d_in[6];
  const float* W2 = (const float*)d_in[7];
  const float* b2 = (const float*)d_in[8];
  const float* Hm = (const float*)d_in[9];
  float* out = (float*)d_out;

  // Workspace carve-up (all vector-access offsets 32B aligned):
  float* H1 = (float*)d_ws;                      // [kN,kH] f32    10.24 MB
  float* Tb = H1 + (size_t)kN * kH;              // [kN,kC] f32  (logits)
  float* Eb = Tb + (size_t)kN * kC;              // [kN,kC] f32  (E_hat)
  float* Bh = Eb + (size_t)kN * kC;              // [kN,kC] f32  (B_hat)
  __bf16* XWt = (__bf16*)(Bh + (size_t)kN * kC); // [kH][kN]  (feat@W1)^T
  __bf16* W1t = XWt + (size_t)kH * kN;           // [kH][kF]
  __bf16* W2t = W1t + (size_t)kH * kF;           // [kC][kH]
  __bf16* Bht = W2t + (size_t)kC * kH;           // [kC][kN]  (H1@W2)^T
  __bf16* Tt  = Bht + (size_t)kC * kN;           // [kC][kN]  (B@H)^T
  float* Pbuf = (float*)(Tt + (size_t)kC * kN);  // [kSplit][kN,kC] partials

  const int mtiles     = kN / 16;                // 625
  const int narrowBlks = (mtiles + 3) / 4;       // 157
  const int rowBlks    = (kN + 255) / 256;
  const int elemBlks   = (kN * kC + 255) / 256;  // 625
  const int ktTotalN   = (kN + 31) / 32;         // 313
  const int ktPer      = (ktTotalN + kSplit - 1) / kSplit;  // 40
  const dim3 splitGrid(narrowBlks, kSplit);      // 1256 blocks = 5024 waves

  // 0) weight transposes to bf16
  transpose_cvt<<<(kF * kH + 255) / 256, 256, 0, stream>>>(W1, W1t, kF, kH);
  transpose_cvt<<<(kH * kC + 255) / 256, 256, 0, stream>>>(W2, W2t, kH, kC);

  // 1) XWt = (feat @ W1)^T  bf16              [10000x512]@[512x256]
  wmma_mm<4, false, false><<<mtiles, 128, 0, stream>>>(
      feat, W1t, nullptr, XWt, kN, kF, kH, 0, nullptr, nullptr, 0.0f);
  // 2) H1 = relu(normed @ XW + b1)            [10000x10000]@[10000x256]
  wmma_mm<4, true, false><<<mtiles, 128, 0, stream>>>(
      normed, XWt, H1, nullptr, kN, kN, kH, 0, b1, nullptr, 0.0f);
  // 3) Bht = (H1 @ W2)^T  bf16                [10000x256]@[256x16]
  wmma_mm<1, false, false><<<narrowBlks, 128, 0, stream>>>(
      H1, W2t, nullptr, Bht, kN, kH, kC, 0, nullptr, nullptr, 0.0f);
  // 4) logits = normed @ (H1@W2) + b2         (k-split + reduce)
  wmma_mm<1, false, true><<<splitGrid, 128, 0, stream>>>(
      normed, Bht, Pbuf, nullptr, kN, kN, kC, ktPer, nullptr, nullptr, 0.0f);
  reduce_split<<<elemBlks, 256, 0, stream>>>(Pbuf, Tb, b2, nullptr, 0.0f, kN);
  // 5) Eb = softmax(logits) - 1/C
  softmax_e<<<rowBlks, 256, 0, stream>>>(Tb, Eb, kN, 1.0f / kC);

  // 6) 4x belief propagation:  B <- E + raw_adj @ (B @ H)
  for (int it = 0; it < kIters; ++it) {
    const float* src = (it == 0) ? Eb : Bh;
    small_mm_t<<<rowBlks, 256, 0, stream>>>(src, Hm, Tt, kN);
    wmma_mm<1, false, true><<<splitGrid, 128, 0, stream>>>(
        raw_adj, Tt, Pbuf, nullptr, kN, kN, kC, ktPer, nullptr, nullptr, 0.0f);
    float* dst  = (it == kIters - 1) ? out : Bh;
    float  sadd = (it == kIters - 1) ? (1.0f / kC) : 0.0f;  // final +1/C
    reduce_split<<<elemBlks, 256, 0, stream>>>(Pbuf, dst, nullptr, Eb, sadd, kN);
  }
}
